// CARAFE_31507880084217
// MI455X (gfx1250) — compile-verified, matrix operand order checked by reading, not run on hardware
//
#include <hip/hip_runtime.h>

// ---------------------------------------------------------------------------
// CARAFE on gfx1250: bf16 WMMA GEMMs for the three convs + VALU reassembly.
//   x (4,256,64,64) f32
//   down: 1x1 conv 256->128        => GEMM [16384,256]x[256,128]   (WMMA bf16)
//   enc : 3x3 conv 128->100        => GEMM [16384,1152]x[1152,112] (WMMA bf16, im2col)
//   pixel_shuffle(2) + softmax(25) + 5x5 reassembly (VALU f32)
//   out : 1x1 conv 256->256 on 128x128 => GEMM [65536,256]x[256,256] (WMMA bf16)
// A-tiles of the 1x1 GEMMs are staged once per block into LDS with
// global_load_async_to_lds_b128 (ASYNCcnt) since all waves share the M-tile.
// ---------------------------------------------------------------------------

typedef __attribute__((ext_vector_type(16))) __bf16 v16bf;
typedef __attribute__((ext_vector_type(8)))  __bf16 v8bf;
typedef __attribute__((ext_vector_type(8)))  float  v8f;

#define NIMG 4
#define CIN  256
#define HH   64
#define WW   64
#define CD   128
#define CE   100
#define CEP  112          // CE padded to 7 WMMA N-tiles
#define KIM  1152         // CD * 9 taps
#define NPIX (NIMG*HH*WW)         // 16384 input pixels
#define NPIXO (NIMG*2*HH*2*WW)    // 65536 output pixels

__device__ __forceinline__ __bf16 f2bf(float f) {
  unsigned u = __float_as_uint(f);
  // round-to-nearest-even bf16
  unsigned r = (u + 0x7FFFu + ((u >> 16) & 1u)) >> 16;
  unsigned short h = (unsigned short)r;
  __bf16 b;
  __builtin_memcpy(&b, &h, 2);
  return b;
}

// Load one lane's 16-half fragment slice. CDNA5 16-bit A/B layout per lane:
// K(e) = kbase + 16*(e>>3) + (e&7), so it is two contiguous 16B runs.
__device__ __forceinline__ v16bf load_frag16(const __bf16* p) {
  v8bf lo = *reinterpret_cast<const v8bf*>(p);
  v8bf hi = *reinterpret_cast<const v8bf*>(p + 16);
  v16bf f;
#pragma unroll
  for (int i = 0; i < 8; ++i) { f[i] = lo[i]; f[i + 8] = hi[i]; }
  return f;
}

__device__ __forceinline__ v8f wmma_bf16(v16bf a, v16bf b, v8f c) {
  return __builtin_amdgcn_wmma_f32_16x16x32_bf16(false, a, false, b, (short)0, c,
                                                 false, false);
}

// Async copy of 16 bytes per lane: global (saddr + voffset) -> LDS (vdst addr).
// GVS addressing mode; tracked by ASYNCcnt.
__device__ __forceinline__ void async_copy_b128(unsigned lds_byte_addr,
                                                unsigned g_byte_off,
                                                unsigned long long g_base) {
  asm volatile("global_load_async_to_lds_b128 %0, %1, %2"
               :: "v"(lds_byte_addr), "v"(g_byte_off), "s"(g_base)
               : "memory");
}

__device__ __forceinline__ void wait_asynccnt0() {
  asm volatile("s_wait_asynccnt 0" ::: "memory");
}

// ---------------------------------------------------------------------------
// Prep kernels: transpose/convert operands into WMMA-friendly bf16 layouts.
// ---------------------------------------------------------------------------

// x (NCHW f32) -> xt [pixel][channel] bf16 (A operand of down-GEMM)
__global__ __launch_bounds__(256) void cvt_x_kernel(const float* __restrict__ x,
                                                    __bf16* __restrict__ xt) {
  int pix = blockIdx.x * 256 + threadIdx.x;   // 0..NPIX-1
  int n  = pix >> 12;                         // /4096
  int hw = pix & 4095;
  const float* px = x + (size_t)n * CIN * 4096 + hw;
  __bf16* po = xt + (size_t)pix * CIN;
#pragma unroll 4
  for (int c = 0; c < CIN; ++c) po[c] = f2bf(px[(size_t)c * 4096]);
}

// w_down [co][ci] f32 -> bf16 same layout (B operand: lane column co, contiguous k)
__global__ __launch_bounds__(256) void cvt_wdown_kernel(const float* __restrict__ w,
                                                        __bf16* __restrict__ o) {
  int i = blockIdx.x * 256 + threadIdx.x;
  if (i < CD * CIN) o[i] = f2bf(w[i]);
}

// w_out [co][ci] f32 -> bf16 same layout
__global__ __launch_bounds__(256) void cvt_wout_kernel(const float* __restrict__ w,
                                                       __bf16* __restrict__ o) {
  int i = blockIdx.x * 256 + threadIdx.x;
  if (i < CIN * CIN) o[i] = f2bf(w[i]);
}

// w_enc [ce][cd][3][3] f32 -> wer [ce][tap*128+cd] bf16, ce padded to 112 w/ zeros
__global__ __launch_bounds__(256) void cvt_wenc_kernel(const float* __restrict__ w,
                                                       __bf16* __restrict__ o) {
  int i = blockIdx.x * 256 + threadIdx.x;
  if (i >= CEP * KIM) return;
  int ce = i / KIM;
  int k  = i - ce * KIM;
  int tap = k >> 7;          // /128
  int cd  = k & 127;
  float v = (ce < CE) ? w[((size_t)(ce * CD + cd)) * 9 + tap] : 0.f;
  o[i] = f2bf(v);
}

// ---------------------------------------------------------------------------
// GEMM 1: down 1x1 conv. 1024 blocks x 256 thr (8 waves); each wave = one
// 16x16 tile: M-tile = blockIdx (shared), N-tile = wave. K = 256.
// A tile (16x256 bf16 = 8KB) staged once per block via async global->LDS.
// ---------------------------------------------------------------------------
__global__ __launch_bounds__(256) void gemm_down_kernel(const __bf16* __restrict__ xt,
                                                        const __bf16* __restrict__ wd,
                                                        const float* __restrict__ bdown,
                                                        __bf16* __restrict__ kt1) {
  __shared__ __bf16 atile[16 * CIN];          // 8 KB
  int tid = threadIdx.x;
  int wave = tid >> 5;
  int lane = tid & 31;
  int m0 = blockIdx.x * 16;

  {
    unsigned long long gb = (unsigned long long)(uintptr_t)(xt + (size_t)m0 * CIN);
    unsigned l0 = (unsigned)(uintptr_t)(&atile[0]);
#pragma unroll
    for (int it = 0; it < 2; ++it) {
      unsigned boff = (unsigned)(tid + it * 256) * 16u;   // 2*256*16B = 8192B
      async_copy_b128(l0 + boff, boff, gb);
    }
    wait_asynccnt0();
  }
  __syncthreads();

  int n0 = wave * 16;
  int koff = (lane >> 4) * 8;
  int ncol = n0 + (lane & 15);
  const __bf16* pa = atile + (size_t)(lane & 15) * CIN + koff;   // LDS
  const __bf16* pb = wd + (size_t)ncol * CIN + koff;             // global
  v8f acc = {};
#pragma unroll
  for (int k0 = 0; k0 < CIN; k0 += 32) {
    v16bf a = load_frag16(pa + k0);
    v16bf b = load_frag16(pb + k0);
    acc = wmma_bf16(a, b, acc);
  }
  float bias = bdown[ncol];
  int mbase = m0 + ((lane >> 4) * 8);
#pragma unroll
  for (int r = 0; r < 8; ++r)
    kt1[(size_t)(mbase + r) * CD + ncol] = f2bf(acc[r] + bias);
}

// ---------------------------------------------------------------------------
// GEMM 2: enc 3x3 conv, implicit im2col. 1024 blocks x 224 thr (7 waves =
// 7 N-tiles covering CE padded to 112). K = 9 taps * 128 = 36 WMMA steps.
// A rows are neighbor-pixel dependent (boundary-zeroed), so direct loads.
// ---------------------------------------------------------------------------
__global__ __launch_bounds__(224) void gemm_enc_kernel(const __bf16* __restrict__ kt1,
                                                       const __bf16* __restrict__ wer,
                                                       const float* __restrict__ benc,
                                                       float* __restrict__ kenc) {
  int wave = threadIdx.x >> 5;   // 0..6
  int lane = threadIdx.x & 31;
  int m0 = blockIdx.x * 16;
  int koff = (lane >> 4) * 8;
  int mrow = m0 + (lane & 15);
  int nimg = mrow >> 12;
  int hw = mrow & 4095;
  int h = hw >> 6, w = hw & 63;
  int ncol = wave * 16 + (lane & 15);
  const __bf16* pb = wer + (size_t)ncol * KIM + koff;
  v8f acc = {};
#pragma unroll
  for (int tap = 0; tap < 9; ++tap) {
    int dy = tap / 3 - 1, dx = tap % 3 - 1;
    int hh = h + dy, ww = w + dx;
    bool ok = (hh >= 0) & (hh < HH) & (ww >= 0) & (ww < WW);
    const __bf16* pa =
        kt1 + ((size_t)((nimg << 12) + hh * WW + ww)) * CD + koff;
#pragma unroll
    for (int cd0 = 0; cd0 < CD; cd0 += 32) {
      v16bf a = {};
      if (ok) a = load_frag16(pa + cd0);
      v16bf b = load_frag16(pb + tap * CD + cd0);
      acc = wmma_bf16(a, b, acc);
    }
  }
  if (ncol < CE) {
    float bias = benc[ncol];
    int mbase = m0 + ((lane >> 4) * 8);
#pragma unroll
    for (int r = 0; r < 8; ++r)
      kenc[(size_t)(mbase + r) * CE + ncol] = acc[r] + bias;
  }
}

// ---------------------------------------------------------------------------
// Softmax + 5x5 reassembly. One block per (n, h) row, 256 threads.
// Phase 1: 64 pixels * 4 subpixels softmax over 25 taps -> LDS.
// Phase 2: per-channel 25-tap weighted sums -> rt [outpixel][channel] bf16.
// ---------------------------------------------------------------------------
__global__ __launch_bounds__(256) void reassemble_kernel(const float* __restrict__ x,
                                                         const float* __restrict__ kenc,
                                                         __bf16* __restrict__ rt) {
  __shared__ float wts[64 * 100];
  int nb = blockIdx.x;           // 0..255 = n*64 + h
  int n = nb >> 6, h = nb & 63;
  int tid = threadIdx.x;

  {
    int w = tid >> 2, sub = tid & 3;
    int pix = (n << 12) + (h << 6) + w;
    const float* p = kenc + (size_t)pix * CE + sub;
    float v[25];
    float mx = -3.4e38f;
#pragma unroll
    for (int k = 0; k < 25; ++k) { v[k] = p[k * 4]; mx = fmaxf(mx, v[k]); }
    float s = 0.f;
#pragma unroll
    for (int k = 0; k < 25; ++k) { v[k] = __expf(v[k] - mx); s += v[k]; }
    float inv = __frcp_rn(s);
#pragma unroll
    for (int k = 0; k < 25; ++k) wts[w * 100 + k * 4 + sub] = v[k] * inv;
  }
  __syncthreads();

  int w = tid & 63, cg = tid >> 6;   // 64 pixels x 4 channel-groups
  const float* wp0 = wts + w * 100;
  for (int cc = 0; cc < CIN / 4; ++cc) {
    int c = cc * 4 + cg;
    const float* xb = x + ((size_t)(n * CIN + c)) * 4096;
    float a0 = 0.f, a1 = 0.f, a2 = 0.f, a3 = 0.f;
#pragma unroll
    for (int k = 0; k < 25; ++k) {
      int dy = k / 5 - 2, dx = k % 5 - 2;
      int hh = h + dy, ww = w + dx;
      float v = ((hh >= 0) & (hh < HH) & (ww >= 0) & (ww < WW))
                    ? xb[hh * WW + ww] : 0.f;
      const float* wp = wp0 + k * 4;
      a0 = fmaf(v, wp[0], a0);
      a1 = fmaf(v, wp[1], a1);
      a2 = fmaf(v, wp[2], a2);
      a3 = fmaf(v, wp[3], a3);
    }
    int y0 = 2 * h, x0 = 2 * w;
    __bf16* r0 = rt + ((size_t)((n * 128 + y0) * 128 + x0)) * CIN + c;
    r0[0]               = f2bf(a0);   // (i=0,j=0)
    r0[CIN]             = f2bf(a1);   // (i=0,j=1)
    r0[128 * CIN]       = f2bf(a2);   // (i=1,j=0)
    r0[128 * CIN + CIN] = f2bf(a3);   // (i=1,j=1)
  }
}

// ---------------------------------------------------------------------------
// GEMM 3: out 1x1 conv on (4,256,128,128). 4096 blocks x 256 thr (8 waves);
// each wave handles 2 N-tiles (A fragment reused). A tile staged via async
// global->LDS (shared across all 8 waves). Output fp32 NCHW.
// ---------------------------------------------------------------------------
__global__ __launch_bounds__(256) void gemm_out_kernel(const __bf16* __restrict__ rt,
                                                       const __bf16* __restrict__ wo,
                                                       const float* __restrict__ bout,
                                                       float* __restrict__ out) {
  __shared__ __bf16 atile[16 * CIN];          // 8 KB
  int tid = threadIdx.x;
  int wave = tid >> 5;
  int lane = tid & 31;
  int m0 = blockIdx.x * 16;

  {
    unsigned long long gb = (unsigned long long)(uintptr_t)(rt + (size_t)m0 * CIN);
    unsigned l0 = (unsigned)(uintptr_t)(&atile[0]);
#pragma unroll
    for (int it = 0; it < 2; ++it) {
      unsigned boff = (unsigned)(tid + it * 256) * 16u;
      async_copy_b128(l0 + boff, boff, gb);
    }
    wait_asynccnt0();
  }
  __syncthreads();

  int koff = (lane >> 4) * 8;
  const __bf16* pa = atile + (size_t)(lane & 15) * CIN + koff;   // LDS
  int n0 = wave * 32;
  int nc0 = n0 + (lane & 15);
  int nc1 = nc0 + 16;
  const __bf16* pb0 = wo + (size_t)nc0 * CIN + koff;
  const __bf16* pb1 = wo + (size_t)nc1 * CIN + koff;
  v8f acc0 = {}, acc1 = {};
#pragma unroll
  for (int k0 = 0; k0 < CIN; k0 += 32) {
    v16bf a  = load_frag16(pa + k0);
    v16bf b0 = load_frag16(pb0 + k0);
    v16bf b1 = load_frag16(pb1 + k0);
    acc0 = wmma_bf16(a, b0, acc0);
    acc1 = wmma_bf16(a, b1, acc1);
  }
  float bias0 = bout[nc0];
  float bias1 = bout[nc1];
  int mbase = m0 + ((lane >> 4) * 8);
#pragma unroll
  for (int r = 0; r < 8; ++r) {
    int m = mbase + r;
    int nimg = m >> 14;          // /16384 (128*128)
    int rem = m & 16383;
    out[((size_t)(nimg * CIN + nc0) << 14) + rem] = acc0[r] + bias0;
    out[((size_t)(nimg * CIN + nc1) << 14) + rem] = acc1[r] + bias1;
  }
}

// ---------------------------------------------------------------------------
extern "C" void kernel_launch(void* const* d_in, const int* in_sizes, int n_in,
                              void* d_out, int out_size, void* d_ws, size_t ws_size,
                              hipStream_t stream) {
  const float* x      = (const float*)d_in[0];
  const float* w_down = (const float*)d_in[1];
  const float* b_down = (const float*)d_in[2];
  const float* w_enc  = (const float*)d_in[3];
  const float* b_enc  = (const float*)d_in[4];
  const float* w_out  = (const float*)d_in[5];
  const float* b_out  = (const float*)d_in[6];
  float* out = (float*)d_out;

  char* ws = (char*)d_ws;
  auto carve = [&](size_t bytes) -> char* {
    char* p = ws;
    ws += (bytes + 255) & ~(size_t)255;
    return p;
  };
  __bf16* xt   = (__bf16*)carve((size_t)NPIX * CIN * 2);   // 8 MB
  __bf16* wd   = (__bf16*)carve((size_t)CD * CIN * 2);     // 64 KB
  __bf16* wer  = (__bf16*)carve((size_t)CEP * KIM * 2);    // 252 KB
  __bf16* wo   = (__bf16*)carve((size_t)CIN * CIN * 2);    // 128 KB
  __bf16* kt1  = (__bf16*)carve((size_t)NPIX * CD * 2);    // 4 MB
  float*  kenc = (float*)carve((size_t)NPIX * CE * 4);     // 6.5 MB
  __bf16* rt   = (__bf16*)carve((size_t)NPIXO * CIN * 2);  // 32 MB

  cvt_x_kernel<<<NPIX / 256, 256, 0, stream>>>(x, xt);
  cvt_wdown_kernel<<<(CD * CIN + 255) / 256, 256, 0, stream>>>(w_down, wd);
  cvt_wenc_kernel<<<(CEP * KIM + 255) / 256, 256, 0, stream>>>(w_enc, wer);
  cvt_wout_kernel<<<(CIN * CIN + 255) / 256, 256, 0, stream>>>(w_out, wo);

  gemm_down_kernel<<<NPIX / 16, 256, 0, stream>>>(xt, wd, b_down, kt1);
  gemm_enc_kernel<<<NPIX / 16, 224, 0, stream>>>(kt1, wer, b_enc, kenc);
  reassemble_kernel<<<NIMG * HH, 256, 0, stream>>>(x, kenc, rt);
  gemm_out_kernel<<<NPIXO / 16, 256, 0, stream>>>(rt, wo, b_out, out);
}
